// MultiHeadAttention_67087389164179
// MI455X (gfx1250) — compile-verified, matrix operand order checked by reading, not run on hardware
//
#include <hip/hip_runtime.h>
#include <hip/hip_bf16.h>
#include <stdint.h>

#define B_   4
#define S_   2048
#define D_   768
#define H_   12
#define DK_  64
#define MTOT (B_ * S_)   // 8192

typedef __attribute__((ext_vector_type(16))) __bf16 v16bf;
typedef __attribute__((ext_vector_type(8)))  float  v8f;

struct __attribute__((aligned(16))) U4 { unsigned int x, y, z, w; };
union Frag16 { v16bf v; U4 u[2]; };

__device__ __forceinline__ v8f wmma_bf16(v16bf a, v16bf b, v8f c) {
  return __builtin_amdgcn_wmma_f32_16x16x32_bf16(
      /*neg_a=*/false, a, /*neg_b=*/false, b,
      /*c_mod=*/(short)0, c, /*reuse_a=*/false, /*reuse_b=*/false);
}

// ---- CDNA5 async global->LDS DMA (ASYNCcnt-tracked, no VGPR round trip) ----
// INST_OFFSET is added to BOTH the LDS dest and global src (ISA 08 §4.4).
__device__ __forceinline__ void asyncStage64B(const __bf16* gsrc, __bf16* ldst) {
  unsigned l = (unsigned)(uintptr_t)ldst;       // flat low 32 bits == LDS offset
  uint64_t g = (uint64_t)(uintptr_t)gsrc;
  asm volatile(
      "global_load_async_to_lds_b128 %0, %1, off\n\t"
      "global_load_async_to_lds_b128 %0, %1, off offset:16\n\t"
      "global_load_async_to_lds_b128 %0, %1, off offset:32\n\t"
      "global_load_async_to_lds_b128 %0, %1, off offset:48"
      :: "v"(l), "v"(g) : "memory");
}
__device__ __forceinline__ void asyncStage32B(const __bf16* gsrc, __bf16* ldst) {
  unsigned l = (unsigned)(uintptr_t)ldst;
  uint64_t g = (uint64_t)(uintptr_t)gsrc;
  asm volatile(
      "global_load_async_to_lds_b128 %0, %1, off\n\t"
      "global_load_async_to_lds_b128 %0, %1, off offset:16"
      :: "v"(l), "v"(g) : "memory");
}
__device__ __forceinline__ void waitAsync() {
  asm volatile("s_wait_asynccnt 0x0" ::: "memory");
}

// Load a 16-bit A/B fragment from LDS (K-contiguous rows).
// lanes 0-15 hold k = {0..7, 16..23}, lanes 16-31 hold k = {8..15, 24..31}.
__device__ __forceinline__ v16bf ldsFragO(const __bf16* base, int row,
                                          int strideBytes, int half, int kByteOff) {
  Frag16 f;
  const char* p = (const char*)base + row * strideBytes + kByteOff + half * 16;
  f.u[0] = *(const U4*)(p);
  f.u[1] = *(const U4*)(p + 32);
  return f.v;
}

// ---------------------------------------------------------------------------
__global__ void cvt_bf16(const float* __restrict__ in, __bf16* __restrict__ out, int n) {
  int i = blockIdx.x * blockDim.x + threadIdx.x;
  int stride = gridDim.x * blockDim.x;
  for (; i < n; i += stride) out[i] = (__bf16)in[i];
}

// Wt[n*D + k] = (bf16) W[k*D + n]
__global__ void transpose_bf16(const float* __restrict__ W, __bf16* __restrict__ Wt) {
  int i = blockIdx.x * blockDim.x + threadIdx.x;
  if (i < D_ * D_) {
    int n = i / D_, k = i % D_;
    Wt[i] = (__bf16)W[(size_t)k * D_ + n];
  }
}

// ---------------------------------------------------------------------------
// C[m,n] = sum_k A[m,k]*Wt[n,k] + bias[n]; 128x128 tile, 4 waves, double-buffered.
// mode 0: (C+bias)*scale -> bf16 head-split [B,H,S,DK]   (Q, K)
// mode 2: (C+bias)*scale -> bf16 transposed [B,H,DK,S]   (V)
// mode 1: (C+bias)       -> f32 row-major [M,D]          (output proj)
#define GK       32
#define LDSTRIDE 48   // elements per LDS row (96B)

__global__ void __launch_bounds__(128)
gemm128(const __bf16* __restrict__ A, const __bf16* __restrict__ Wt,
        const float* __restrict__ bias,
        __bf16* __restrict__ outHS, float* __restrict__ outF,
        float scale, int mode) {
  __shared__ __bf16 ldsA[2][128 * LDSTRIDE];
  __shared__ __bf16 ldsB[2][128 * LDSTRIDE];

  const int tid  = threadIdx.x;
  const int lane = tid & 31, wave = tid >> 5;
  const int lr   = lane & 15, half = lane >> 4;
  const int wrow = wave >> 1, wcol = wave & 1;
  const int m0 = blockIdx.x * 128;
  const int n0 = blockIdx.y * 128;

  v8f acc[4][4];
#pragma unroll
  for (int r = 0; r < 4; ++r)
#pragma unroll
    for (int c = 0; c < 4; ++c)
#pragma unroll
      for (int i = 0; i < 8; ++i) acc[r][c][i] = 0.0f;

  // Prologue: stage first K-slab
  asyncStage64B(A  + (size_t)(m0 + tid) * D_, ldsA[0] + tid * LDSTRIDE);
  asyncStage64B(Wt + (size_t)(n0 + tid) * D_, ldsB[0] + tid * LDSTRIDE);
  waitAsync();
  __syncthreads();

  for (int k0 = 0; k0 < D_; k0 += GK) {
    const int buf = (k0 / GK) & 1;
    if (k0 + GK < D_) {   // overlap next slab's DMA with this slab's WMMAs
      asyncStage64B(A  + (size_t)(m0 + tid) * D_ + k0 + GK, ldsA[buf ^ 1] + tid * LDSTRIDE);
      asyncStage64B(Wt + (size_t)(n0 + tid) * D_ + k0 + GK, ldsB[buf ^ 1] + tid * LDSTRIDE);
    }

    v16bf af[4], bfr[4];
#pragma unroll
    for (int r = 0; r < 4; ++r)
      af[r] = ldsFragO(ldsA[buf], wrow * 64 + r * 16 + lr, LDSTRIDE * 2, half, 0);
#pragma unroll
    for (int c = 0; c < 4; ++c)
      bfr[c] = ldsFragO(ldsB[buf], wcol * 64 + c * 16 + lr, LDSTRIDE * 2, half, 0);
#pragma unroll
    for (int r = 0; r < 4; ++r)
#pragma unroll
      for (int c = 0; c < 4; ++c)
        acc[r][c] = wmma_bf16(af[r], bfr[c], acc[r][c]);

    waitAsync();
    __syncthreads();
  }

  // Epilogue. C layout: vgpr i, lanes 0-15 -> (M=i, N=lane); 16-31 -> (M=8+i)
#pragma unroll
  for (int c = 0; c < 4; ++c) {
    int n = n0 + wcol * 64 + c * 16 + lr;
    float bv = bias[n];
#pragma unroll
    for (int r = 0; r < 4; ++r) {
#pragma unroll
      for (int i = 0; i < 8; ++i) {
        int m = m0 + wrow * 64 + r * 16 + i + half * 8;
        float val = (acc[r][c][i] + bv) * scale;
        if (mode == 0) {
          int b = m / S_, s = m % S_;
          int h = n / DK_, dk = n % DK_;
          outHS[(((size_t)b * H_ + h) * S_ + s) * DK_ + dk] = (__bf16)val;
        } else if (mode == 2) {
          int b = m / S_, s = m % S_;
          int h = n / DK_, dk = n % DK_;
          outHS[(((size_t)b * H_ + h) * DK_ + dk) * S_ + s] = (__bf16)val;
        } else {
          outF[(size_t)m * D_ + n] = val;
        }
      }
    }
  }
}

// ---------------------------------------------------------------------------
// Flash-style attention. V arrives pre-transposed [B,H,DK,S] so both K and V
// tiles stage via async b128 DMA, double-buffered.
#define KSTR 72   // ldsK row stride (elems) = 144B
#define VSTR 48   // ldsVt row stride (elems) = 96B
#define PSTR 48   // ldsP  row stride (elems) = 96B

__global__ void __launch_bounds__(128)
attn(const __bf16* __restrict__ Qh, const __bf16* __restrict__ Kh,
     const __bf16* __restrict__ Vt, const unsigned char* __restrict__ mask,
     __bf16* __restrict__ ctx) {
  __shared__ __bf16 ldsK[2][32 * KSTR];    // row = key, K-contig dk
  __shared__ __bf16 ldsVt[2][64 * VSTR];   // row = dk, cols = keys (pre-transposed)
  __shared__ __bf16 ldsP[4 * 16 * PSTR];   // per-wave P re-layout buffer

  const int tid  = threadIdx.x;
  const int lane = tid & 31, wave = tid >> 5;
  const int lr   = lane & 15, half = lane >> 4;
  const int b = blockIdx.z, h = blockIdx.y;
  const int bh = b * H_ + h;
  const int q0 = blockIdx.x * 64 + wave * 16;

  const __bf16* Krow = Kh + (size_t)bh * S_ * DK_;   // [key][dk]
  const __bf16* Vrow = Vt + (size_t)bh * DK_ * S_;   // [dk][key]

  // Per-thread staging roles
  const int skey = tid >> 2, sch = tid & 3;   // K: 32 keys x 4 chunks(16 elems)
  const int sdk  = tid >> 1, vch = tid & 1;   // V: 64 dk rows x 2 chunks(16 elems)

  // Q fragments (16 queries x DK=64 -> two 16x32 A fragments), pre-scaled by 1/8
  v16bf qf[2];
  {
    const char* rowp = (const char*)(Qh + ((size_t)bh * S_ + q0 + lr) * DK_);
#pragma unroll
    for (int ks = 0; ks < 2; ++ks) {
      Frag16 f;
      f.u[0] = *(const U4*)(rowp + ks * 64 + half * 16);
      f.u[1] = *(const U4*)(rowp + ks * 64 + half * 16 + 32);
      qf[ks] = f.v;
    }
  }

  v8f accv[4];
#pragma unroll
  for (int j = 0; j < 4; ++j)
#pragma unroll
    for (int i = 0; i < 8; ++i) accv[j][i] = 0.0f;

  float mi[8], li[8];
#pragma unroll
  for (int i = 0; i < 8; ++i) { mi[i] = -3.0e38f; li[i] = 0.0f; }

  const unsigned char* mrow = mask + (size_t)b * S_;

  // Prologue: stage first key tile
  asyncStage32B(Krow + (size_t)skey * DK_ + sch * 16, ldsK[0] + skey * KSTR + sch * 16);
  asyncStage32B(Vrow + (size_t)sdk * S_ + vch * 16,   ldsVt[0] + sdk * VSTR + vch * 16);
  waitAsync();
  __syncthreads();

  for (int kb0 = 0; kb0 < S_; kb0 += 32) {
    const int buf = (kb0 >> 5) & 1;
    if (kb0 + 32 < S_) {   // overlap next tile's DMA with this tile's math
      asyncStage32B(Krow + (size_t)(kb0 + 32 + skey) * DK_ + sch * 16,
                    ldsK[buf ^ 1] + skey * KSTR + sch * 16);
      asyncStage32B(Vrow + (size_t)sdk * S_ + kb0 + 32 + vch * 16,
                    ldsVt[buf ^ 1] + sdk * VSTR + vch * 16);
    }

    // Scores: two 16x16 fragments (key groups), K=64 in two WMMA steps
    v8f s[2];
#pragma unroll
    for (int g = 0; g < 2; ++g) {
      v8f c;
#pragma unroll
      for (int i = 0; i < 8; ++i) c[i] = 0.0f;
      c = wmma_bf16(qf[0], ldsFragO(ldsK[buf], g * 16 + lr, KSTR * 2, half, 0), c);
      c = wmma_bf16(qf[1], ldsFragO(ldsK[buf], g * 16 + lr, KSTR * 2, half, 64), c);
      s[g] = c;
    }

    // Mask (per-lane column): True = masked position
#pragma unroll
    for (int g = 0; g < 2; ++g) {
      int mk = mrow[kb0 + g * 16 + lr];
      if (mk) {
#pragma unroll
        for (int i = 0; i < 8; ++i) s[g][i] = -1.0e30f;
      }
    }

    // Online softmax (row lives across 16 lanes of a half-wave)
    float p0[8], p1[8];
#pragma unroll
    for (int i = 0; i < 8; ++i) {
      float t = fmaxf(s[0][i], s[1][i]);
#pragma unroll
      for (int off = 1; off <= 8; off <<= 1)
        t = fmaxf(t, __shfl_xor(t, off, 32));
      float mn = fmaxf(mi[i], t);
      float corr = __expf(mi[i] - mn);
      mi[i] = mn;
      p0[i] = __expf(s[0][i] - mn);
      p1[i] = __expf(s[1][i] - mn);
      float rs = p0[i] + p1[i];
#pragma unroll
      for (int off = 1; off <= 8; off <<= 1)
        rs += __shfl_xor(rs, off, 32);
      li[i] = li[i] * corr + rs;
#pragma unroll
      for (int j = 0; j < 4; ++j) accv[j][i] *= corr;
    }

    // Re-layout P (C layout -> A layout) through per-wave LDS region
    __bf16* pb = ldsP + wave * 16 * PSTR;
#pragma unroll
    for (int i = 0; i < 8; ++i) {
      int m = i + half * 8;
      pb[m * PSTR + lr]      = (__bf16)p0[i];
      pb[m * PSTR + 16 + lr] = (__bf16)p1[i];
    }
    asm volatile("s_wait_dscnt 0x0" ::: "memory");  // wave-local LDS visibility

    v16bf pf = ldsFragO(pb, lr, PSTR * 2, half, 0);
#pragma unroll
    for (int j = 0; j < 4; ++j) {
      v16bf vf = ldsFragO(ldsVt[buf], j * 16 + lr, VSTR * 2, half, 0);
      accv[j] = wmma_bf16(pf, vf, accv[j]);
    }

    waitAsync();
    __syncthreads();
  }

  // Normalize and write context in [B,S,D] layout
#pragma unroll
  for (int j = 0; j < 4; ++j)
#pragma unroll
    for (int i = 0; i < 8; ++i) {
      int q = q0 + i + half * 8;
      float val = accv[j][i] / li[i];
      ctx[((size_t)b * S_ + q) * D_ + h * DK_ + j * 16 + lr] = (__bf16)val;
    }
}

// ---------------------------------------------------------------------------
extern "C" void kernel_launch(void* const* d_in, const int* in_sizes, int n_in,
                              void* d_out, int out_size, void* d_ws, size_t ws_size,
                              hipStream_t stream) {
  (void)in_sizes; (void)n_in; (void)out_size; (void)ws_size;
  const float* q  = (const float*)d_in[0];
  const float* k  = (const float*)d_in[1];
  const float* v  = (const float*)d_in[2];
  const unsigned char* mask = (const unsigned char*)d_in[3];
  const float* Wq = (const float*)d_in[4];
  const float* bq = (const float*)d_in[5];
  const float* Wk = (const float*)d_in[6];
  const float* bk = (const float*)d_in[7];
  const float* Wv = (const float*)d_in[8];
  const float* bv = (const float*)d_in[9];
  const float* Wo = (const float*)d_in[10];
  const float* bo = (const float*)d_in[11];
  float* out = (float*)d_out;

  const size_t actN = (size_t)MTOT * D_;
  const size_t wN   = (size_t)D_ * D_;
  __bf16* ws  = (__bf16*)d_ws;
  __bf16* qb  = ws;  ws += actN;
  __bf16* kb  = ws;  ws += actN;
  __bf16* vb  = ws;  ws += actN;
  __bf16* wtq = ws;  ws += wN;
  __bf16* wtk = ws;  ws += wN;
  __bf16* wtv = ws;  ws += wN;
  __bf16* wto = ws;  ws += wN;
  __bf16* Qh  = ws;  ws += actN;
  __bf16* Kh  = ws;  ws += actN;
  __bf16* Vt  = ws;  ws += actN;
  __bf16* ctx = ws;  ws += actN;

  cvt_bf16<<<1024, 256, 0, stream>>>(q, qb, (int)actN);
  cvt_bf16<<<1024, 256, 0, stream>>>(k, kb, (int)actN);
  cvt_bf16<<<1024, 256, 0, stream>>>(v, vb, (int)actN);
  transpose_bf16<<<(D_ * D_ + 255) / 256, 256, 0, stream>>>(Wq, wtq);
  transpose_bf16<<<(D_ * D_ + 255) / 256, 256, 0, stream>>>(Wk, wtk);
  transpose_bf16<<<(D_ * D_ + 255) / 256, 256, 0, stream>>>(Wv, wtv);
  transpose_bf16<<<(D_ * D_ + 255) / 256, 256, 0, stream>>>(Wo, wto);

  dim3 gg(MTOT / 128, D_ / 128);
  gemm128<<<gg, 128, 0, stream>>>(qb, wtq, bq, Qh, nullptr, 0.125f, 0);  // Q * 1/sqrt(DK)
  gemm128<<<gg, 128, 0, stream>>>(kb, wtk, bk, Kh, nullptr, 1.0f, 0);
  gemm128<<<gg, 128, 0, stream>>>(vb, wtv, bv, Vt, nullptr, 1.0f, 2);    // V transposed

  attn<<<dim3(S_ / 64, H_, B_), 128, 0, stream>>>(Qh, Kh, Vt, mask, ctx);

  gemm128<<<gg, 128, 0, stream>>>(ctx, wto, bo, nullptr, out, 1.0f, 1);
}
